// GAT1NN2CMLRV_7567732376160
// MI455X (gfx1250) — compile-verified, matrix operand order checked by reading, not run on hardware
//
#include <hip/hip_runtime.h>
#include <hip/hip_bf16.h>

// ---- problem constants (from reference) ----
#define PB      256   // batch
#define PN      64    // nodes
#define PT      64    // time
#define PH      256   // in features
#define NFEAT   256
#define NHID    64
#define NHEADS  8
#define POUT    512
#define KBIG    (PN * NHEADS * NHID)   // 32768
#define MROWS   (PB * PN)              // 16384
#define KSPLIT  8                      // K-chunks for the final GEMM

typedef __attribute__((ext_vector_type(16))) __bf16 v16bf;
typedef __attribute__((ext_vector_type(8)))  __bf16 v8bf;
typedef __attribute__((ext_vector_type(4)))  __bf16 v4bf;
typedef __attribute__((ext_vector_type(8)))  float  v8f;

// ---- fp32 <-> bf16 helpers (bit-level, RNE) ----
__device__ __forceinline__ __bf16 f2bf(float f) {
    unsigned u = __builtin_bit_cast(unsigned, f);
    unsigned r = (u + 0x7FFFu + ((u >> 16) & 1u)) >> 16;
    unsigned short s = (unsigned short)r;
    return __builtin_bit_cast(__bf16, s);
}
__device__ __forceinline__ float bf2f(__bf16 b) {
    unsigned short s = __builtin_bit_cast(unsigned short, b);
    unsigned u = ((unsigned)s) << 16;
    return __builtin_bit_cast(float, u);
}

// load 16 bf16 for one WMMA operand: two contiguous 16B chunks at p and p+16
__device__ __forceinline__ v16bf ld_op16(const __bf16* p) {
    v8bf lo = *(const v8bf*)p;
    v8bf hi = *(const v8bf*)(p + 16);
    v16bf r;
#pragma unroll
    for (int j = 0; j < 8; ++j) { r[j] = lo[j]; r[j + 8] = hi[j]; }
    return r;
}

__device__ __forceinline__ v8f wmma_bf16(v16bf a, v16bf b, v8f c) {
    return __builtin_amdgcn_wmma_f32_16x16x32_bf16(
        /*neg_a=*/false, a, /*neg_b=*/false, b,
        /*c_mod=*/(short)0, c, /*reuse_a=*/false, /*reuse_b=*/false);
}

// ------------------------------------------------------------------
// K0: unpad lengths from x[0, :, :, 0]
// ------------------------------------------------------------------
__global__ void unpad_len_kernel(const float* __restrict__ x, int* lenI, float* invL) {
    int n = threadIdx.x;            // 64 threads
    if (n >= PN) return;
    const float* p = x + (size_t)n * PT * PH;   // batch 0, feature 0
    int last = -1;
#pragma unroll 4
    for (int t = 0; t < PT; ++t)
        if (p[(size_t)t * PH] != 0.0f) last = t;
    int len = (last >= 0) ? (last + 1) : PT;
    lenI[n] = len;
    invL[n] = 1.0f / (float)len;
}

// ------------------------------------------------------------------
// K1: masked mean over T -> bf16 (MROWS x PH).  Streams 1.07 GB of x at the
// HBM roofline (~46us @ 23.3 TB/s).  float4 lanes => global_load_b128,
// 512B per wave-instruction; 4 t-groups per block reduced through LDS.
// ------------------------------------------------------------------
__global__ void mean_kernel(const float* __restrict__ x, const int* __restrict__ lenI,
                            const float* __restrict__ invL, __bf16* __restrict__ meanBf) {
    const int bn  = blockIdx.x;          // 0..16383  (b*64+n)
    const int n   = bn & (PN - 1);
    const int tid = threadIdx.x;         // 256
    const int c4  = tid & 63;            // float4 column 0..63
    const int tg  = tid >> 6;            // t-group 0..3
    const float4* p = (const float4*)(x + (size_t)bn * PT * PH) + c4;
    const int len = lenI[n];

    float sx = 0.f, sy = 0.f, sz = 0.f, sw = 0.f;
    for (int t = tg; t < len; t += 4) {
        if (t + 8 < len) __builtin_prefetch(p + (size_t)(t + 8) * (PH / 4), 0, 0);
        float4 v = p[(size_t)t * (PH / 4)];
        sx += v.x; sy += v.y; sz += v.z; sw += v.w;
    }

    __shared__ float red[4][256];
    red[0][tid] = sx; red[1][tid] = sy; red[2][tid] = sz; red[3][tid] = sw;
    __syncthreads();

    if (tg == 0) {
        const float il = invL[n];
        v4bf o;
#pragma unroll
        for (int comp = 0; comp < 4; ++comp) {
            float s = red[comp][c4] + red[comp][c4 + 64] +
                      red[comp][c4 + 128] + red[comp][c4 + 192];
            o[comp] = f2bf(s * il);
        }
        *(v4bf*)(meanBf + (size_t)bn * PH + c4 * 4) = o;   // 8B store
    }
}

// ------------------------------------------------------------------
// K2: tiled transpose fp32(R x C) -> bf16(C x R), batched via blockIdx.z
// ------------------------------------------------------------------
__global__ void transpose_f32_bf16(const float* __restrict__ in, __bf16* __restrict__ out,
                                   int R, int C, size_t inBatchStride, size_t outBatchStride) {
    __shared__ float tile[32][33];
    const float* ip = in + (size_t)blockIdx.z * inBatchStride;
    __bf16* op = out + (size_t)blockIdx.z * outBatchStride;
    int c0 = blockIdx.x * 32, r0 = blockIdx.y * 32;
    int tx = threadIdx.x, ty = threadIdx.y;     // 32 x 8
#pragma unroll
    for (int k = 0; k < 32; k += 8) {
        int r = r0 + ty + k, c = c0 + tx;
        if (r < R && c < C) tile[ty + k][tx] = ip[(size_t)r * C + c];
    }
    __syncthreads();
#pragma unroll
    for (int k = 0; k < 32; k += 8) {
        int orow = c0 + ty + k;   // output row = input col
        int ocol = r0 + tx;       // output col = input row
        if (orow < C && ocol < R) op[(size_t)orow * R + ocol] = f2bf(tile[tx][ty + k]);
    }
}

// ------------------------------------------------------------------
// Register-blocked WMMA GEMM:  out(M x N) = A(M x K) * Bt(N x K)^T (+bias)
// 32x32 per wave (2x2 tiles of 16x16): 4 WMMAs per 4 operand loads -> 2x
// register reuse.  8 waves/block.  M,N multiples of 32; K multiple of 32.
// ------------------------------------------------------------------
__global__ void gemm_bf16_wmma_2x2(const __bf16* __restrict__ A, const __bf16* __restrict__ Bt,
                                   __bf16* __restrict__ outBf, const float* __restrict__ bias,
                                   int M, int N, int K) {
    const int wave = threadIdx.x >> 5;
    const int lane = threadIdx.x & 31;
    const int r    = lane & 15;
    const int half = lane >> 4;
    const int tilesN2 = N >> 5;
    const int tile = blockIdx.x * 8 + wave;
    const int tm2 = tile / tilesN2, tn2 = tile % tilesN2;
    if (tm2 * 32 >= M) return;

    const __bf16* a0 = A  + (size_t)(tm2 * 32 + r) * K + half * 8;
    const __bf16* a1 = a0 + (size_t)16 * K;
    const __bf16* b0 = Bt + (size_t)(tn2 * 32 + r) * K + half * 8;
    const __bf16* b1 = b0 + (size_t)16 * K;

    v8f acc00 = {}, acc01 = {}, acc10 = {}, acc11 = {};
#pragma unroll 2
    for (int k0 = 0; k0 < K; k0 += 32) {
        v16bf av0 = ld_op16(a0 + k0);
        v16bf av1 = ld_op16(a1 + k0);
        v16bf bv0 = ld_op16(b0 + k0);
        v16bf bv1 = ld_op16(b1 + k0);
        acc00 = wmma_bf16(av0, bv0, acc00);
        acc01 = wmma_bf16(av0, bv1, acc01);
        acc10 = wmma_bf16(av1, bv0, acc10);
        acc11 = wmma_bf16(av1, bv1, acc11);
    }

    const int c0 = tn2 * 32 + r, c1 = c0 + 16;
    const float bb0 = bias ? bias[c0] : 0.0f;
    const float bb1 = bias ? bias[c1] : 0.0f;
#pragma unroll
    for (int j = 0; j < 8; ++j) {
        int r0 = tm2 * 32 + half * 8 + j, r1 = r0 + 16;
        outBf[(size_t)r0 * N + c0] = f2bf(acc00[j] + bb0);
        outBf[(size_t)r0 * N + c1] = f2bf(acc01[j] + bb1);
        outBf[(size_t)r1 * N + c0] = f2bf(acc10[j] + bb0);
        outBf[(size_t)r1 * N + c1] = f2bf(acc11[j] + bb1);
    }
}

// ------------------------------------------------------------------
// K-split WMMA GEMM for small-M/huge-K: each wave does a 16x16 tile over a
// K/KSPLIT chunk and accumulates into fp32 scratch with HW f32 atomics.
// Raises wave count 8x for the 256x512x32768 GEMM.
// ------------------------------------------------------------------
__global__ void gemm_bf16_wmma_ksplit(const __bf16* __restrict__ A, const __bf16* __restrict__ Bt,
                                      float* __restrict__ part, int M, int N, int K) {
    const int wave = threadIdx.x >> 5;
    const int lane = threadIdx.x & 31;
    const int r    = lane & 15;
    const int half = lane >> 4;
    const int tilesN = N >> 4;
    const int tilesPerK = (M >> 4) * tilesN;
    const int gid  = blockIdx.x * 8 + wave;
    const int kc   = gid / tilesPerK;
    const int tile = gid % tilesPerK;
    const int tm = tile / tilesN, tn = tile % tilesN;
    const int kLen  = K / KSPLIT;
    const int kBase = kc * kLen;
    if (kc >= KSPLIT) return;

    const __bf16* arow = A  + (size_t)(tm * 16 + r) * K + kBase + half * 8;
    const __bf16* brow = Bt + (size_t)(tn * 16 + r) * K + kBase + half * 8;

    v8f acc = {};
#pragma unroll 4
    for (int k0 = 0; k0 < kLen; k0 += 32) {
        v16bf av = ld_op16(arow + k0);
        v16bf bv = ld_op16(brow + k0);
        acc = wmma_bf16(av, bv, acc);
    }

    const int col = tn * 16 + r;
#pragma unroll
    for (int j = 0; j < 8; ++j) {
        int row = tm * 16 + half * 8 + j;
        unsafeAtomicAdd(&part[(size_t)row * N + col], acc[j]);   // global_atomic_add_f32
    }
}

__global__ void zero_kernel(float* __restrict__ p, int n) {
    int i = blockIdx.x * 256 + threadIdx.x;
    if (i < n) p[i] = 0.0f;
}

__global__ void bias_relu_kernel(const float* __restrict__ part, const float* __restrict__ bias,
                                 float* __restrict__ out, int n) {
    int i = blockIdx.x * 256 + threadIdx.x;
    if (i < n) {
        float v = part[i] + bias[i & (POUT - 1)];
        out[i] = v > 0.0f ? v : 0.0f;
    }
}

// ------------------------------------------------------------------
// K5: fused GAT attention per (head, batch): f1/f2 dots, leaky-relu,
// softmax, att@Wh via WMMA, elu.  Wh laid out (b*64+n, head*64+h) bf16.
// ------------------------------------------------------------------
__global__ void gat_attention_kernel(const __bf16* __restrict__ Wh,
                                     const float* __restrict__ att_a,
                                     __bf16* __restrict__ g) {
    const int head = blockIdx.x;    // 0..7
    const int b    = blockIdx.y;    // 0..255
    __shared__ __attribute__((aligned(16))) __bf16 sWh [PN * NHID];  // (n,h)
    __shared__ __attribute__((aligned(16))) __bf16 sWhT[NHID * PN];  // (h,m)  B-operand
    __shared__ __attribute__((aligned(16))) __bf16 sAtt[PN * PN];    // (n,m)  A-operand
    __shared__ float sF1[PN], sF2[PN], sA1[NHID], sA2[NHID];

    const int tid = threadIdx.x;    // 256

    for (int idx = tid; idx < PN * NHID; idx += 256) {
        int n = idx >> 6, hh = idx & 63;
        __bf16 v = Wh[(size_t)(b * PN + n) * (NHEADS * NHID) + head * NHID + hh];
        sWh[idx] = v;
        sWhT[hh * PN + n] = v;
    }
    if (tid < NHID) {
        sA1[tid] = att_a[head * 2 * NHID + tid];
        sA2[tid] = att_a[head * 2 * NHID + NHID + tid];
    }
    __syncthreads();

    if (tid < PN) {                 // f1[n], f2[n]
        float s1 = 0.0f, s2 = 0.0f;
#pragma unroll 4
        for (int hh = 0; hh < NHID; ++hh) {
            float w = bf2f(sWh[tid * NHID + hh]);
            s1 += w * sA1[hh];
            s2 += w * sA2[hh];
        }
        sF1[tid] = s1; sF2[tid] = s2;
    }
    __syncthreads();

    if (tid < PN) {                 // softmax row tid over m
        const float f1 = sF1[tid];
        float mx = -3.4e38f;
        for (int m = 0; m < PN; ++m) {
            float e = f1 + sF2[m]; e = e > 0.0f ? e : 0.2f * e;
            mx = e > mx ? e : mx;
        }
        float sum = 0.0f;
        for (int m = 0; m < PN; ++m) {
            float e = f1 + sF2[m]; e = e > 0.0f ? e : 0.2f * e;
            sum += __expf(e - mx);
        }
        float inv = 1.0f / sum;
        for (int m = 0; m < PN; ++m) {
            float e = f1 + sF2[m]; e = e > 0.0f ? e : 0.2f * e;
            sAtt[tid * PN + m] = f2bf(__expf(e - mx) * inv);
        }
    }
    __syncthreads();

    // att(64x64) @ Wh(64x64): 16 tiles of 16x16, 2 per wave, K = 64
    const int wave = tid >> 5, lane = tid & 31;
    const int r = lane & 15, half = lane >> 4;
#pragma unroll
    for (int it = 0; it < 2; ++it) {
        int tile = wave * 2 + it;
        int tm = tile >> 2, tn = tile & 3;
        const __bf16* arow = sAtt + (tm * 16 + r) * PN + half * 8;
        const __bf16* brow = sWhT + (tn * 16 + r) * PN + half * 8;
        v8f acc = {};
#pragma unroll
        for (int k0 = 0; k0 < PN; k0 += 32) {
            acc = wmma_bf16(ld_op16(arow + k0), ld_op16(brow + k0), acc);
        }
        int col = tn * 16 + r;
#pragma unroll
        for (int j = 0; j < 8; ++j) {
            int row = tm * 16 + half * 8 + j;
            float v = acc[j];
            v = v > 0.0f ? v : (__expf(v) - 1.0f);   // elu
            g[(size_t)(b * PN + row) * (NHEADS * NHID) + head * NHID + col] = f2bf(v);
        }
    }
}

// ------------------------------------------------------------------
// host launcher
// ------------------------------------------------------------------
extern "C" void kernel_launch(void* const* d_in, const int* in_sizes, int n_in,
                              void* d_out, int out_size, void* d_ws, size_t ws_size,
                              hipStream_t stream) {
    const float* x       = (const float*)d_in[0];   // (256,64,64,256)
    const float* share_w = (const float*)d_in[1];   // (256,256)
    const float* share_b = (const float*)d_in[2];   // (256,)
    const float* att_w   = (const float*)d_in[3];   // (8,256,64)
    const float* att_a   = (const float*)d_in[4];   // (8,128,1)
    const float* lin1_w  = (const float*)d_in[5];   // (32768,512)
    const float* lin1_b  = (const float*)d_in[6];   // (512,)
    float* out = (float*)d_out;                     // (256,512)

    // workspace carve (~82 MB total)
    char* w = (char*)d_ws;
    auto carve = [&](size_t bytes) -> char* {
        char* p = w;
        w += (bytes + 255) & ~(size_t)255;
        return p;
    };
    int*    lenI   = (int*)   carve(PN * sizeof(int));
    float*  invL   = (float*) carve(PN * sizeof(float));
    __bf16* meanBf = (__bf16*)carve((size_t)MROWS * PH * 2);                 // 8 MB
    __bf16* swT    = (__bf16*)carve((size_t)NFEAT * PH * 2);                 // 128 KB
    __bf16* attT   = (__bf16*)carve((size_t)NHEADS * NHID * NFEAT * 2);      // 256 KB
    __bf16* linT   = (__bf16*)carve((size_t)POUT * KBIG * 2);                // 32 MB
    __bf16* hBf    = (__bf16*)carve((size_t)MROWS * NFEAT * 2);              // 8 MB
    __bf16* WhBf   = (__bf16*)carve((size_t)MROWS * NHEADS * NHID * 2);      // 16 MB
    __bf16* gBf    = (__bf16*)carve((size_t)MROWS * NHEADS * NHID * 2);      // 16 MB
    float*  partF  = (float*) carve((size_t)PB * POUT * sizeof(float));      // 512 KB
    (void)ws_size; (void)n_in; (void)in_sizes; (void)out_size;

    // K0: unpad lengths
    unpad_len_kernel<<<1, 64, 0, stream>>>(x, lenI, invL);

    // K1: masked mean -> bf16 (HBM-bound floor)
    mean_kernel<<<MROWS, 256, 0, stream>>>(x, lenI, invL, meanBf);

    // K2: weight transposes/convert to K-contiguous bf16 layouts
    transpose_f32_bf16<<<dim3(PH / 32, PH / 32, 1), dim3(32, 8), 0, stream>>>(
        share_w, swT, PH, NFEAT, 0, 0);
    transpose_f32_bf16<<<dim3(NHID / 32, NFEAT / 32, NHEADS), dim3(32, 8), 0, stream>>>(
        att_w, attT, NFEAT, NHID, (size_t)NFEAT * NHID, (size_t)NHID * NFEAT);
    transpose_f32_bf16<<<dim3(POUT / 32, KBIG / 32, 1), dim3(32, 8), 0, stream>>>(
        lin1_w, linT, KBIG, POUT, 0, 0);

    // K3: h = mean @ share_w + share_b       (16384 x 256 x 256)
    gemm_bf16_wmma_2x2<<<(MROWS / 32) * (NFEAT / 32) / 8, 256, 0, stream>>>(
        meanBf, swT, hBf, share_b, MROWS, NFEAT, PH);

    // K4: Wh (all 8 heads fused) = h @ attT^T  (16384 x 512 x 256)
    gemm_bf16_wmma_2x2<<<(MROWS / 32) * (NHEADS * NHID / 32) / 8, 256, 0, stream>>>(
        hBf, attT, WhBf, nullptr, MROWS, NHEADS * NHID, NFEAT);

    // K5: per-(head,batch) attention + elu -> g
    gat_attention_kernel<<<dim3(NHEADS, PB), 256, 0, stream>>>(WhBf, att_a, gBf);

    // K6: out = relu(g @ lin1_w + lin1_b)    (256 x 512 x 32768), K-split
    {
        const int nOut = PB * POUT;
        zero_kernel<<<(nOut + 255) / 256, 256, 0, stream>>>(partF, nOut);
        const int tiles = (PB / 16) * (POUT / 16) * KSPLIT;   // 4096 wave-tiles
        gemm_bf16_wmma_ksplit<<<tiles / 8, 256, 0, stream>>>(
            gBf, linT, partF, PB, POUT, KBIG);
        bias_relu_kernel<<<(nOut + 255) / 256, 256, 0, stream>>>(partF, lin1_b, out, nOut);
    }
}
